// Decoder_67954972557669
// MI455X (gfx1250) — matrix operand order repaired and checked by reading, not compiled, not run on hardware
//
#include <hip/hip_runtime.h>
#include <hip/hip_bf16.h>

// ---------------------------------------------------------------------------
// Problem dimensions (fixed by the reference)
// ---------------------------------------------------------------------------
#define BB   64
#define SS   256
#define TT   128
#define HH   1024
#define EE   512
#define H2   2048      // 2H
#define H3   3072      // 3H
#define XK   2560      // E + 2H  (GRU input width)
#define CK   3584      // E + H + 2H (pre-output concat width)

typedef __attribute__((ext_vector_type(16))) __bf16 v16bf;
typedef __attribute__((ext_vector_type(8)))  __bf16 v8bf;
typedef __attribute__((ext_vector_type(8)))  float  v8f;
typedef __attribute__((ext_vector_type(4)))  float  f32x4;

static __device__ __forceinline__ __bf16 f2bf(float x) { return (__bf16)x; }

// ---------------------------------------------------------------------------
// WMMA fragment layouts (ISA 7.12.2, wave32):
//   A 16x32 (MxK): lane l -> row l&15; K chunks [8h,8h+8) u [16+8h,16+8h+8)
//   B 32x16 (KxN) from row-major W[N,K]: lane l -> col l&15; K chunks [8h,8h+8) u [16+8h,16+8h+8)
//   C/D f32: VGPR i -> row i+8h, col l&15
// ---------------------------------------------------------------------------
static __device__ __forceinline__ v16bf a_frag_from(const __bf16* __restrict__ arow,
                                                    int k0) {
    v8bf lo = *reinterpret_cast<const v8bf*>(arow + k0);
    v8bf hi = *reinterpret_cast<const v8bf*>(arow + k0 + 16);
    v16bf f;
#pragma unroll
    for (int j = 0; j < 8; ++j) { f[j] = lo[j]; f[j + 8] = hi[j]; }
    return f;
}
static __device__ __forceinline__ v16bf b_frag_from(const __bf16* __restrict__ brow,
                                                    int k0) {
    v8bf lo = *reinterpret_cast<const v8bf*>(brow + k0);
    v8bf hi = *reinterpret_cast<const v8bf*>(brow + k0 + 16);
    v16bf f;
#pragma unroll
    for (int j = 0; j < 8; ++j) { f[j] = lo[j]; f[j + 8] = hi[j]; }
    return f;
}

// ---------------------------------------------------------------------------
// GEMM: C[M,N] = act( A[M,K] (bf16) * W[N,K]^T (bf16) + bias ), f32 accumulate
//   Register-blocked: each wave owns MT(4) m-tiles x NT(2) n-tiles = 64x32 C,
//   8 independent WMMAs per K-step (no D->A/B hazards -> no NOP padding).
//   grid: (N/128, M/64, nbatch), block = 128 (4 waves, each a distinct n-slab)
// ---------------------------------------------------------------------------
#define GEMM_MT 4
#define GEMM_NT 2
#define PF_DIST 512   // prefetch distance in K elements (1 KB of bf16)

__global__ __launch_bounds__(128)
void wmma_gemm_kernel(const __bf16* __restrict__ A, const __bf16* __restrict__ W,
                      const float* __restrict__ bias, float* __restrict__ C,
                      int M, int N, int K, int lda,
                      long batchStrideA, long batchStrideC, int ldc, int act) {
    const int lane = threadIdx.x & 31;
    const int wave = threadIdx.x >> 5;
    const int h = lane >> 4, r = lane & 15;
    const int n0 = (blockIdx.x * 4 + wave) * (16 * GEMM_NT);
    const int m0 = blockIdx.y * (16 * GEMM_MT);

    A += (size_t)blockIdx.z * batchStrideA;
    C += (size_t)blockIdx.z * batchStrideC;

    // hoisted per-lane row pointers
    const __bf16* arow[GEMM_MT];
#pragma unroll
    for (int mt = 0; mt < GEMM_MT; ++mt)
        arow[mt] = A + (size_t)(m0 + mt * 16 + r) * lda + 8 * h;
    const __bf16* brow[GEMM_NT];
#pragma unroll
    for (int nt = 0; nt < GEMM_NT; ++nt)
        brow[nt] = W + (size_t)(n0 + nt * 16 + r) * K + 8 * h;

    v8f acc[GEMM_MT][GEMM_NT];
#pragma unroll
    for (int mt = 0; mt < GEMM_MT; ++mt)
#pragma unroll
        for (int nt = 0; nt < GEMM_NT; ++nt)
            acc[mt][nt] = (v8f){0.f, 0.f, 0.f, 0.f, 0.f, 0.f, 0.f, 0.f};

    for (int k0 = 0; k0 < K; k0 += 32) {
        v16bf b[GEMM_NT];
#pragma unroll
        for (int nt = 0; nt < GEMM_NT; ++nt) b[nt] = b_frag_from(brow[nt], k0);
        v16bf a[GEMM_MT];
#pragma unroll
        for (int mt = 0; mt < GEMM_MT; ++mt) a[mt] = a_frag_from(arow[mt], k0);
#pragma unroll
        for (int mt = 0; mt < GEMM_MT; ++mt)
#pragma unroll
            for (int nt = 0; nt < GEMM_NT; ++nt)
                acc[mt][nt] = __builtin_amdgcn_wmma_f32_16x16x32_bf16(
                    false, a[mt], false, b[nt], (short)0, acc[mt][nt], false, false);
        if (k0 + PF_DIST < K) {               // keep L2 weight stream ahead
#pragma unroll
            for (int nt = 0; nt < GEMM_NT; ++nt)
                __builtin_prefetch((const void*)(brow[nt] + k0 + PF_DIST), 0, 3);
        }
    }

#pragma unroll
    for (int nt = 0; nt < GEMM_NT; ++nt) {
        const int col = n0 + nt * 16 + r;
        const float bv = bias ? bias[col] : 0.f;
#pragma unroll
        for (int mt = 0; mt < GEMM_MT; ++mt) {
#pragma unroll
            for (int i = 0; i < 8; ++i) {
                const int row = m0 + mt * 16 + i + 8 * h;
                float v = acc[mt][nt][i] + bv;
                if (act == 1) v = tanhf(v);
                C[(size_t)row * ldc + col] = v;
            }
        }
    }
}

// ---------------------------------------------------------------------------
// f32 -> bf16 bulk convert (weights + encoder tensors, once per launch)
// ---------------------------------------------------------------------------
__global__ __launch_bounds__(256)
void cvt_bf16_kernel(const float* __restrict__ in, __bf16* __restrict__ out, long n) {
    long i = (long)blockIdx.x * blockDim.x + threadIdx.x;
    if (i < n) out[i] = f2bf(in[i]);
}

// ---------------------------------------------------------------------------
// energies[b,s] = sum_h v[h] * tanh(q[b,h] + proj_key[b,s,h]); masked
//   one wave per (b,s) row; float4 per lane (b128 loads); grid = B*S/8
// ---------------------------------------------------------------------------
__global__ __launch_bounds__(256)
void energies_kernel(const float* __restrict__ q, const float* __restrict__ pk,
                     const float* __restrict__ v, const unsigned char* __restrict__ mask,
                     float* __restrict__ e) {
    const int lane = threadIdx.x & 31;
    const int wave = threadIdx.x >> 5;
    const int idx  = blockIdx.x * 8 + wave;    // flat b*S + s
    const int b    = idx >> 8;                 // S = 256
    const f32x4* q4 = reinterpret_cast<const f32x4*>(q + (size_t)b * HH);
    const f32x4* p4 = reinterpret_cast<const f32x4*>(pk + (size_t)idx * HH);
    const f32x4* v4 = reinterpret_cast<const f32x4*>(v);
    float acc = 0.f;
#pragma unroll
    for (int it = 0; it < HH / 128; ++it) {    // 8 iterations
        const int o = it * 32 + lane;
        f32x4 qa = q4[o], pa = p4[o], va = v4[o];
#pragma unroll
        for (int j = 0; j < 4; ++j) acc += va[j] * tanhf(qa[j] + pa[j]);
    }
#pragma unroll
    for (int o = 16; o > 0; o >>= 1) acc += __shfl_xor(acc, o, 32);
    if (lane == 0) {
        const int s = idx & (SS - 1);
        e[idx] = mask[(size_t)b * SS + s] ? acc : -1.0e9f;
    }
}

// ---------------------------------------------------------------------------
// per-b: softmax over S=256, then context[b,:] = alphas @ encoder_hidden[b]
// writes bf16 context into the GRU-input buffer x and the pre-concat buffer
//   grid = B, block = 256; each thread owns 8 contiguous d's (2x b128 loads)
// ---------------------------------------------------------------------------
__global__ __launch_bounds__(256)
void softmax_ctx_kernel(const float* __restrict__ e, const float* __restrict__ enc,
                        __bf16* __restrict__ xbuf, __bf16* __restrict__ catbuf) {
    __shared__ float sh[256];
    __shared__ float alphas[256];
    const int b = blockIdx.x, tid = threadIdx.x;

    const float val = e[(size_t)b * SS + tid];
    sh[tid] = val; __syncthreads();
    for (int o = 128; o > 0; o >>= 1) {
        if (tid < o) sh[tid] = fmaxf(sh[tid], sh[tid + o]);
        __syncthreads();
    }
    const float m = sh[0]; __syncthreads();
    const float ex = __expf(val - m);
    sh[tid] = ex; __syncthreads();
    for (int o = 128; o > 0; o >>= 1) {
        if (tid < o) sh[tid] += sh[tid + o];
        __syncthreads();
    }
    alphas[tid] = ex / sh[0];
    __syncthreads();

    const float* eb = enc + (size_t)b * SS * H2 + tid * 8;   // 8 contiguous d
    float acc[8];
#pragma unroll
    for (int i = 0; i < 8; ++i) acc[i] = 0.f;
    for (int s = 0; s < SS; ++s) {
        const float al = alphas[s];
        const f32x4* r4 = reinterpret_cast<const f32x4*>(eb + (size_t)s * H2);
        f32x4 r0 = r4[0], r1 = r4[1];
#pragma unroll
        for (int j = 0; j < 4; ++j) { acc[j] += al * r0[j]; acc[4 + j] += al * r1[j]; }
    }
    __bf16* xo = xbuf + (size_t)b * XK + EE + tid * 8;
    __bf16* co = catbuf + (size_t)b * CK + EE + HH + tid * 8;
#pragma unroll
    for (int i = 0; i < 8; ++i) { xo[i] = f2bf(acc[i]); co[i] = f2bf(acc[i]); }
}

// copy prev_embed (step t) into x and cat buffers (bf16); grid covers B*E
__global__ __launch_bounds__(256)
void embed_copy_kernel(const float* __restrict__ trg, __bf16* __restrict__ xbuf,
                       __bf16* __restrict__ catbuf, int t) {
    const int i = blockIdx.x * blockDim.x + threadIdx.x;  // B*E = 32768
    const int b = i >> 9, ee = i & (EE - 1);
    const __bf16 v = f2bf(trg[((size_t)b * TT + t) * EE + ee]);
    xbuf[(size_t)b * XK + ee]  = v;
    catbuf[(size_t)b * CK + ee] = v;
}

// ---------------------------------------------------------------------------
// GRU elementwise update: h_new = (1-z)*n + z*h
// writes: h (f32, in place), h_bf, cat_bf[:, E:E+H], decoder_states[:, t, :]
// ---------------------------------------------------------------------------
__global__ __launch_bounds__(256)
void gru_update_kernel(const float* __restrict__ gi, const float* __restrict__ gh,
                       float* __restrict__ h, __bf16* __restrict__ h_bf,
                       __bf16* __restrict__ catbuf,
                       float* __restrict__ out_states, int t) {
    const int i = blockIdx.x * blockDim.x + threadIdx.x;  // B*H = 65536
    const int b = i >> 10, hh = i & (HH - 1);
    const float* gib = gi + (size_t)b * H3;
    const float* ghb = gh + (size_t)b * H3;
    const float ir = gib[hh], iz = gib[HH + hh], in_ = gib[2 * HH + hh];
    const float hr = ghb[hh], hz = ghb[HH + hh], hn  = ghb[2 * HH + hh];
    const float r = 1.f / (1.f + __expf(-(ir + hr)));
    const float z = 1.f / (1.f + __expf(-(iz + hz)));
    const float n = tanhf(in_ + r * hn);
    const float hnew = (1.f - z) * n + z * h[i];
    h[i] = hnew;
    h_bf[i] = f2bf(hnew);
    catbuf[(size_t)b * CK + EE + hh] = f2bf(hnew);
    out_states[((size_t)b * TT + t) * HH + hh] = hnew;
}

__global__ __launch_bounds__(256)
void copy_f32_kernel(const float* __restrict__ src, float* __restrict__ dst, long n) {
    long i = (long)blockIdx.x * blockDim.x + threadIdx.x;
    if (i < n) dst[i] = src[i];
}

// ---------------------------------------------------------------------------
// Host driver
// ---------------------------------------------------------------------------
extern "C" void kernel_launch(void* const* d_in, const int* in_sizes, int n_in,
                              void* d_out, int out_size, void* d_ws, size_t ws_size,
                              hipStream_t stream) {
    (void)in_sizes; (void)n_in; (void)out_size; (void)ws_size;

    const float*         trg_embed  = (const float*)d_in[0];   // [B,T,E]
    const float*         enc_hidden = (const float*)d_in[1];   // [B,S,2H]
    const float*         enc_final  = (const float*)d_in[2];   // [1,B,2H]
    const unsigned char* src_mask   = (const unsigned char*)d_in[3]; // [B,1,S] bool
    // d_in[4] = trg_mask (unused by the reference)
    const float* W_bridge = (const float*)d_in[5];
    const float* b_bridge = (const float*)d_in[6];
    const float* W_key    = (const float*)d_in[7];
    const float* W_query  = (const float*)d_in[8];
    const float* v_energy = (const float*)d_in[9];
    const float* W_ih     = (const float*)d_in[10];
    const float* W_hh     = (const float*)d_in[11];
    const float* b_ih     = (const float*)d_in[12];
    const float* b_hh     = (const float*)d_in[13];
    const float* W_pre    = (const float*)d_in[14];

    float* out_states = (float*)d_out;                              // [B,T,H]
    float* out_hidden = out_states + (size_t)BB * TT * HH;          // [1,B,H]
    float* out_pre    = out_hidden + (size_t)BB * HH;               // [B,T,H]

    // -------- workspace bump allocation (256B aligned) --------
    char* wp = (char*)d_ws;
    auto alloc = [&](size_t bytes) -> void* {
        void* p = (void*)wp;
        wp += (bytes + 255) & ~(size_t)255;
        return p;
    };
    __bf16* wbridge  = (__bf16*)alloc((size_t)HH * H2 * 2);
    __bf16* wkey     = (__bf16*)alloc((size_t)HH * H2 * 2);
    __bf16* wquery   = (__bf16*)alloc((size_t)HH * HH * 2);
    __bf16* wih      = (__bf16*)alloc((size_t)H3 * XK * 2);
    __bf16* whh      = (__bf16*)alloc((size_t)H3 * HH * 2);
    __bf16* wpre     = (__bf16*)alloc((size_t)HH * CK * 2);
    __bf16* ench_bf  = (__bf16*)alloc((size_t)BB * SS * H2 * 2);    // 67 MB
    __bf16* encf_bf  = (__bf16*)alloc((size_t)BB * H2 * 2);
    float*  proj_key = (float*)alloc((size_t)BB * SS * HH * 4);     // 67 MB
    float*  h        = (float*)alloc((size_t)BB * HH * 4);
    __bf16* h_bf     = (__bf16*)alloc((size_t)BB * HH * 2);
    float*  q        = (float*)alloc((size_t)BB * HH * 4);
    float*  energ    = (float*)alloc((size_t)BB * SS * 4);
    __bf16* x_bf     = (__bf16*)alloc((size_t)BB * XK * 2);
    __bf16* cat_bf   = (__bf16*)alloc((size_t)BB * CK * 2);
    float*  gi       = (float*)alloc((size_t)BB * H3 * 4);
    float*  gh       = (float*)alloc((size_t)BB * H3 * 4);

    auto cvt = [&](const float* src, __bf16* dst, long n) {
        cvt_bf16_kernel<<<dim3((unsigned)((n + 255) / 256)), 256, 0, stream>>>(src, dst, n);
    };
    const int NCOLS = 16 * GEMM_NT * 4;   // columns per block (128)
    auto gemm = [&](const __bf16* A, const __bf16* W, const float* bias,
                    float* C, int M, int N, int K, int lda,
                    long bA, long bC, int ldc, int act, int nbatch) {
        dim3 g((unsigned)(N / NCOLS), (unsigned)(M / (16 * GEMM_MT)),
               (unsigned)nbatch);
        wmma_gemm_kernel<<<g, 128, 0, stream>>>(A, W, bias, C, M, N, K, lda,
                                                bA, bC, ldc, act);
    };

    // -------- one-time setup: convert weights + encoder tensors --------
    cvt(W_bridge,   wbridge, (long)HH * H2);
    cvt(W_key,      wkey,    (long)HH * H2);
    cvt(W_query,    wquery,  (long)HH * HH);
    cvt(W_ih,       wih,     (long)H3 * XK);
    cvt(W_hh,       whh,     (long)H3 * HH);
    cvt(W_pre,      wpre,    (long)HH * CK);
    cvt(enc_hidden, ench_bf, (long)BB * SS * H2);
    cvt(enc_final,  encf_bf, (long)BB * H2);

    // h0 = tanh(encoder_final @ W_bridge^T + b_bridge)   [64,1024]
    gemm(encf_bf, wbridge, b_bridge, h, BB, HH, H2, H2, 0, 0, HH, /*tanh*/1, 1);
    cvt(h, h_bf, (long)BB * HH);   // overwritten each step by gru kernel

    // proj_key[b] = encoder_hidden[b] @ W_key^T   (batched over b)
    gemm(ench_bf, wkey, nullptr, proj_key, SS, HH, H2, H2,
         (long)SS * H2, (long)SS * HH, HH, 0, BB);

    // -------- sequential decode --------
    for (int t = 0; t < TT; ++t) {
        // q = h @ W_query^T
        gemm(h_bf, wquery, nullptr, q, BB, HH, HH, HH, 0, 0, HH, 0, 1);
        // stage prev_embed into x and cat (bf16)
        embed_copy_kernel<<<dim3(BB * EE / 256), 256, 0, stream>>>(
            trg_embed, x_bf, cat_bf, t);
        // energies + mask
        energies_kernel<<<dim3(BB * SS / 8), 256, 0, stream>>>(
            q, proj_key, v_energy, src_mask, energ);
        // softmax + context -> x[:, E:], cat[:, E+H:]
        softmax_ctx_kernel<<<dim3(BB), 256, 0, stream>>>(
            energ, enc_hidden, x_bf, cat_bf);
        // gi = x @ W_ih^T + b_ih ; gh = h @ W_hh^T + b_hh
        gemm(x_bf, wih, b_ih, gi, BB, H3, XK, XK, 0, 0, H3, 0, 1);
        gemm(h_bf, whh, b_hh, gh, BB, H3, HH, HH, 0, 0, H3, 0, 1);
        // GRU update: h, h_bf, cat[:, E:E+H], decoder_states[:, t, :]
        gru_update_kernel<<<dim3(BB * HH / 256), 256, 0, stream>>>(
            gi, gh, h, h_bf, cat_bf, out_states, t);
        // pre[:, t, :] = cat @ W_pre^T   (strided store into output)
        gemm(cat_bf, wpre, nullptr, out_pre + (size_t)t * HH,
             BB, HH, CK, CK, 0, 0, TT * HH, 0, 1);
    }

    // hidden = h_final[None]
    copy_f32_kernel<<<dim3(BB * HH / 256), 256, 0, stream>>>(
        h, out_hidden, (long)BB * HH);
}
